// GATBlock_10368051052623
// MI455X (gfx1250) — matrix operand order repaired and checked by reading, not compiled, not run on hardware
//
#include <hip/hip_runtime.h>

typedef __attribute__((ext_vector_type(16))) _Float16 v16h;
typedef __attribute__((ext_vector_type(8)))  _Float16 v8h;
typedef __attribute__((ext_vector_type(8)))  float    v8f;

#define N_NODES   50000
#define N_EDGES   800000
#define EN        (N_EDGES + N_NODES)   // edges + self loops = 850000
#define IN_CH     128
#define OUT_CH    64
#define HEADS     4
#define HC        (HEADS * OUT_CH)      // 256
#define NEG_SLOPE 0.2f

// ---- monotone uint key for float atomic-max ----
__device__ __forceinline__ unsigned flt_key(float f) {
  unsigned u = __float_as_uint(f);
  return (u & 0x80000000u) ? ~u : (u | 0x80000000u);
}
__device__ __forceinline__ float key_flt(unsigned m) {
  unsigned u = (m & 0x80000000u) ? (m & 0x7FFFFFFFu) : ~m;
  return __uint_as_float(u);
}

// ---- f32 -> f16 conversion ----
__global__ __launch_bounds__(256) void f32_to_f16(const float* __restrict__ in,
                                                  _Float16* __restrict__ outp, int n) {
  int i = blockIdx.x * 256 + threadIdx.x;
  if (i < n) outp[i] = (_Float16)in[i];
}

// ---- init accumulators: out=0, amax=key(-inf)~0, denom=0 ----
__global__ __launch_bounds__(256) void gat_init(float* __restrict__ out,
                                                unsigned* __restrict__ amax,
                                                float* __restrict__ denom) {
  int i = blockIdx.x * 256 + threadIdx.x;        // 50000 blocks -> 12.8M exact
  out[i] = 0.f;
  if (i < N_NODES * HEADS) { amax[i] = 0u; denom[i] = 0.f; }
}

// ---- WMMA GEMM: xl[50000,256] = xh[50000,128] * wh[256,128]^T (NT) ----
// one wave per 16x16 tile; 3125 M-tiles x 16 N-tiles = 50000 waves exact
__global__ __launch_bounds__(256) void gat_gemm_wmma(const _Float16* __restrict__ xh,
                                                     const _Float16* __restrict__ wh,
                                                     float* __restrict__ xl) {
  const int lane   = threadIdx.x & 31;
  const int gw     = blockIdx.x * 8 + (threadIdx.x >> 5);
  const int mt     = gw >> 4;          // 0..3124
  const int nt     = gw & 15;          // 0..15
  const int hi     = lane >> 4;        // 0 / 1 (half-wave)
  const int l16    = lane & 15;
  const int arow   = mt * 16 + l16;    // A: M = lane&15
  const int aklo   = hi * 8;           // lanes16-31 carry K=8..15 / 24..31
  const int bcol   = nt * 16 + l16;    // B: N = lane&15
  const int bklo   = hi * 8;           // lanes16-31 carry K=8..15 / 24..31

  v8f acc = {};
#pragma unroll
  for (int k0 = 0; k0 < IN_CH; k0 += 32) {
    v8h a0 = *(const v8h*)(xh + arow * IN_CH + k0 + aklo);
    v8h a1 = *(const v8h*)(xh + arow * IN_CH + k0 + aklo + 16);
    v8h b0 = *(const v8h*)(wh + bcol * IN_CH + k0 + bklo);
    v8h b1 = *(const v8h*)(wh + bcol * IN_CH + k0 + bklo + 16);
    v16h a, b;
#pragma unroll
    for (int i = 0; i < 8; ++i) { a[i] = a0[i]; a[i + 8] = a1[i];
                                  b[i] = b0[i]; b[i + 8] = b1[i]; }
    acc = __builtin_amdgcn_wmma_f32_16x16x32_f16(false, a, false, b,
                                                 (short)0, acc, false, false);
  }
  const int rbase = hi * 8;            // C/D: lanes16-31 hold M=8..15
#pragma unroll
  for (int r = 0; r < 8; ++r)
    xl[(mt * 16 + rbase + r) * HC + bcol] = acc[r];
}

// ---- per-(node,head) attention logits: a = <xl[n,h,:], att[h,:]> ----
// one wave per (n,h); 200000 waves exact
__global__ __launch_bounds__(256) void gat_attn_logits(const float* __restrict__ xl,
                                                       const float* __restrict__ att_src,
                                                       const float* __restrict__ att_dst,
                                                       float* __restrict__ a_src,
                                                       float* __restrict__ a_dst) {
  const int lane = threadIdx.x & 31;
  const int gw   = blockIdx.x * 8 + (threadIdx.x >> 5);
  const int n    = gw >> 2;
  const int h    = gw & 3;
  const float* row = xl + n * HC + h * OUT_CH;
  float v0 = row[lane], v1 = row[lane + 32];
  float s = v0 * att_src[h * OUT_CH + lane] + v1 * att_src[h * OUT_CH + lane + 32];
  float d = v0 * att_dst[h * OUT_CH + lane] + v1 * att_dst[h * OUT_CH + lane + 32];
#pragma unroll
  for (int off = 16; off; off >>= 1) {
    s += __shfl_down(s, off, 32);
    d += __shfl_down(d, off, 32);
  }
  if (lane == 0) { a_src[gw] = s; a_dst[gw] = d; }
}

// ---- pass 1: alpha = leakyrelu(a_src[src]+a_dst[dst]); segment max ----
__global__ __launch_bounds__(256) void gat_edge_max(const int* __restrict__ ei,
                                                    const float* __restrict__ a_src,
                                                    const float* __restrict__ a_dst,
                                                    float* __restrict__ alpha,
                                                    unsigned* __restrict__ amax) {
  int e = blockIdx.x * 256 + threadIdx.x;
  if (e >= EN) return;
  int s = (e < N_EDGES) ? ei[e]           : (e - N_EDGES);
  int d = (e < N_EDGES) ? ei[N_EDGES + e] : (e - N_EDGES);
#pragma unroll
  for (int h = 0; h < HEADS; ++h) {
    float v = a_src[s * HEADS + h] + a_dst[d * HEADS + h];
    v = (v > 0.f) ? v : NEG_SLOPE * v;
    alpha[e * HEADS + h] = v;
    atomicMax(&amax[d * HEADS + h], flt_key(v));
  }
}

// ---- pass 2: e = exp(alpha - amax[dst]); segment sum ----
__global__ __launch_bounds__(256) void gat_edge_exp(const int* __restrict__ ei,
                                                    float* __restrict__ alpha,
                                                    const unsigned* __restrict__ amax,
                                                    float* __restrict__ denom) {
  int e = blockIdx.x * 256 + threadIdx.x;
  if (e >= EN) return;
  int d = (e < N_EDGES) ? ei[N_EDGES + e] : (e - N_EDGES);
#pragma unroll
  for (int h = 0; h < HEADS; ++h) {
    float m  = key_flt(amax[d * HEADS + h]);
    float ev = __expf(alpha[e * HEADS + h] - m);
    alpha[e * HEADS + h] = ev;
    atomicAdd(&denom[d * HEADS + h], ev);
  }
}

// ---- pass 3: out[dst] += (e/denom[dst]) * xl[src] ; one wave per edge ----
__global__ __launch_bounds__(256) void gat_edge_scatter(const int* __restrict__ ei,
                                                        const float* __restrict__ ev,
                                                        const float* __restrict__ denom,
                                                        const float* __restrict__ xl,
                                                        float* __restrict__ out) {
  const int lane = threadIdx.x & 31;
  const int e    = blockIdx.x * 8 + (threadIdx.x >> 5);   // 106250*8 = 850000 exact
  int s = (e < N_EDGES) ? ei[e]           : (e - N_EDGES);
  int d = (e < N_EDGES) ? ei[N_EDGES + e] : (e - N_EDGES);
  float coef[HEADS];
#pragma unroll
  for (int h = 0; h < HEADS; ++h)
    coef[h] = ev[e * HEADS + h] / denom[d * HEADS + h];
  const float* srow = xl + s * HC;
  float*       drow = out + d * HC;
#pragma unroll
  for (int j = 0; j < 8; ++j) {
    int i = lane + 32 * j;               // channel; head = i/64 = j>>1
    atomicAdd(&drow[i], coef[j >> 1] * srow[i]);
  }
}

// ---- finalize: out = elu(out + bias) ----
__global__ __launch_bounds__(256) void gat_bias_elu(float* __restrict__ out,
                                                    const float* __restrict__ bias) {
  int i = blockIdx.x * 256 + threadIdx.x;   // 50000 blocks -> 12.8M exact
  float v = out[i] + bias[i & (HC - 1)];
  out[i] = (v > 0.f) ? v : (__expf(v) - 1.f);
}

extern "C" void kernel_launch(void* const* d_in, const int* in_sizes, int n_in,
                              void* d_out, int out_size, void* d_ws, size_t ws_size,
                              hipStream_t stream) {
  const float* x       = (const float*)d_in[0];
  const int*   ei      = (const int*)  d_in[1];
  const float* W       = (const float*)d_in[2];
  const float* att_src = (const float*)d_in[3];
  const float* att_dst = (const float*)d_in[4];
  const float* bias    = (const float*)d_in[5];
  float* out = (float*)d_out;

  // workspace layout (bytes)
  char* ws = (char*)d_ws;
  _Float16* xh    = (_Float16*)(ws);                 // 50000*128*2 = 12,800,000
  _Float16* wh    = (_Float16*)(ws + 12800000);      // 256*128*2   =     65,536
  float*    xl    = (float*)   (ws + 12865536);      // 50000*256*4 = 51,200,000
  float*    a_src = (float*)   (ws + 64065536);      // 200000*4
  float*    a_dst = (float*)   (ws + 64865536);      // 200000*4
  unsigned* amax  = (unsigned*)(ws + 65665536);      // 200000*4
  float*    denom = (float*)   (ws + 66465536);      // 200000*4
  float*    ealp  = (float*)   (ws + 67265536);      // 850000*4*4 = 13,600,000

  f32_to_f16<<<25000, 256, 0, stream>>>(x, xh, N_NODES * IN_CH);
  f32_to_f16<<<128,   256, 0, stream>>>(W, wh, HC * IN_CH);
  gat_init<<<50000, 256, 0, stream>>>(out, amax, denom);
  gat_gemm_wmma<<<6250, 256, 0, stream>>>(xh, wh, xl);
  gat_attn_logits<<<25000, 256, 0, stream>>>(xl, att_src, att_dst, a_src, a_dst);
  gat_edge_max<<<(EN + 255) / 256, 256, 0, stream>>>(ei, a_src, a_dst, ealp, amax);
  gat_edge_exp<<<(EN + 255) / 256, 256, 0, stream>>>(ei, ealp, amax, denom);
  gat_edge_scatter<<<EN / 8, 256, 0, stream>>>(ei, ealp, denom, xl, out);
  gat_bias_elu<<<50000, 256, 0, stream>>>(out, bias);
}
